// MLA_15719580303903
// MI455X (gfx1250) — matrix operand order repaired and checked by reading, not compiled, not run on hardware
//
#include <hip/hip_runtime.h>
#include <hip/hip_bf16.h>

typedef __bf16 bf16_t;
typedef __attribute__((ext_vector_type(4)))  __bf16 v4bf;
typedef __attribute__((ext_vector_type(8)))  __bf16 v8bf;
typedef __attribute__((ext_vector_type(16))) __bf16 v16bf;
typedef __attribute__((ext_vector_type(8)))  float  v8f;

#define DEV static __device__ __forceinline__

// ---------------------------------------------------------------------------
// WMMA helper: D = A(16x32 bf16) * B(32x16 bf16) + C(16x16 f32)
// ---------------------------------------------------------------------------
DEV v8f wmma_bf16(v16bf a, v16bf b, v8f c) {
  return __builtin_amdgcn_wmma_f32_16x16x32_bf16(
      /*neg_a=*/false, a, /*neg_b=*/false, b,
      /*c_mod=*/(short)0, c, /*reuse_a=*/false, /*reuse_b=*/false);
}

// A-fragment per-lane layout (16-bit A 16x32): lane holds K chunks [off,off+8)
// and [off+16,off+24) where off = (lane>>4)*8.  Caller passes p already at
// rowBase + off; we read 8 contiguous + 8 contiguous at +16.
DEV v16bf frag_gap(const bf16_t* p) {
  v8bf lo = *(const v8bf*)(p);
  v8bf hi = *(const v8bf*)(p + 16);
  v16bf r;
#pragma unroll
  for (int i = 0; i < 8; ++i) { r[i] = lo[i]; r[i + 8] = hi[i]; }
  return r;
}

// B-fragment in this code's contiguous-k order (A uses it too): lane holds contiguous K range
// of 16 at off = (lane>>4)*16 for column n = lane&15.  Caller passes p at
// colBase + off; we read 16 contiguous bf16.
DEV v16bf frag_contig(const bf16_t* p) {
  v8bf lo = *(const v8bf*)(p);
  v8bf hi = *(const v8bf*)(p + 8);
  v16bf r;
#pragma unroll
  for (int i = 0; i < 8; ++i) { r[i] = lo[i]; r[i + 8] = hi[i]; }
  return r;
}

// ---------------------------------------------------------------------------
// f32 -> bf16 conversion (vectorized x4)
// ---------------------------------------------------------------------------
__global__ __launch_bounds__(256) void cvt_f32_bf16(
    const float* __restrict__ in, bf16_t* __restrict__ out, int n) {
  int i = (blockIdx.x * 256 + threadIdx.x) * 4;
  if (i + 4 <= n) {
    float4 v = *(const float4*)(in + i);
    v4bf o;
    o[0] = (bf16_t)v.x; o[1] = (bf16_t)v.y;
    o[2] = (bf16_t)v.z; o[3] = (bf16_t)v.w;
    *(v4bf*)(out + i) = o;
  }
}

// ---------------------------------------------------------------------------
// TN GEMM:  C[M,N] = A[M,K](bf16) @ W[N,K](bf16)^T   (nn.Linear semantics)
// Wave tile 32(M) x 64(N), 8 waves/block -> block tile 64 x 256.
// MODE 0: store bf16 row-major (M,N)
// MODE 1: store f32  row-major (M,N)
// MODE 2: store bf16 transposed per batch: out[b][n][t], b=m/Tdim, t=m%Tdim
// ---------------------------------------------------------------------------
template <int MODE>
__global__ __launch_bounds__(256) void gemm_tn(
    const bf16_t* __restrict__ A, const bf16_t* __restrict__ W,
    void* __restrict__ Cout, int M, int N, int K, int Tdim) {
  const int lane  = threadIdx.x & 31;
  const int wave  = threadIdx.x >> 5;
  const int lrow  = lane & 15;
  const int lhalf = lane >> 4;
  const int m0 = blockIdx.x * 64  + (wave & 1) * 32;
  const int n0 = blockIdx.y * 256 + (wave >> 1) * 64;

  const bf16_t* arow0 = A + (size_t)(m0 + lrow) * K + lhalf * 16;
  const bf16_t* arow1 = arow0 + (size_t)16 * K;
  const bf16_t* brow[4];
#pragma unroll
  for (int t = 0; t < 4; ++t)
    brow[t] = W + (size_t)(n0 + t * 16 + lrow) * K + lhalf * 16;

  v8f acc[2][4];
#pragma unroll
  for (int mt = 0; mt < 2; ++mt)
#pragma unroll
    for (int nt = 0; nt < 4; ++nt)
#pragma unroll
      for (int r = 0; r < 8; ++r) acc[mt][nt][r] = 0.0f;

  for (int k = 0; k < K; k += 32) {
    v16bf a0 = frag_contig(arow0 + k);
    v16bf a1 = frag_contig(arow1 + k);
#pragma unroll
    for (int t = 0; t < 4; ++t) {
      v16bf bt = frag_contig(brow[t] + k);
      acc[0][t] = wmma_bf16(a0, bt, acc[0][t]);
      acc[1][t] = wmma_bf16(a1, bt, acc[1][t]);
    }
  }

  // C layout: VGPR r, lanes 0-15 -> M=r, N=lane; lanes 16-31 -> M=r+8.
#pragma unroll
  for (int mt = 0; mt < 2; ++mt) {
    const int mbase = m0 + mt * 16 + lhalf * 8;
#pragma unroll
    for (int nt = 0; nt < 4; ++nt) {
      const int n = n0 + nt * 16 + lrow;
      if constexpr (MODE == 1) {
        float* C = (float*)Cout;
#pragma unroll
        for (int r = 0; r < 8; ++r)
          C[(size_t)(mbase + r) * N + n] = acc[mt][nt][r];
      } else if constexpr (MODE == 0) {
        bf16_t* C = (bf16_t*)Cout;
#pragma unroll
        for (int r = 0; r < 8; ++r)
          C[(size_t)(mbase + r) * N + n] = (bf16_t)acc[mt][nt][r];
      } else {  // MODE == 2: transposed (B, N, Tdim); mbase..mbase+7 same batch
        const int bb = mbase / Tdim;
        const int tt = mbase - bb * Tdim;
        bf16_t* C = (bf16_t*)Cout + ((size_t)bb * N + n) * Tdim + tt;
        v8bf pk;
#pragma unroll
        for (int r = 0; r < 8; ++r) pk[r] = (bf16_t)acc[mt][nt][r];
        *(v8bf*)C = pk;  // contiguous along t -> single 16B store
      }
    }
  }
}

// ---------------------------------------------------------------------------
// Flash attention with sliding window (512) + sink (4).
// One wave owns a 16-query tile; 8 waves/block = 128 queries of one (b,h).
// q: (B,T,C) bf16   k: (B,T,C) bf16   vT: (B,C,T) bf16   y: (B,T,C) bf16
// ---------------------------------------------------------------------------
__global__ __launch_bounds__(256) void attn_kernel(
    const bf16_t* __restrict__ qb, const bf16_t* __restrict__ kb,
    const bf16_t* __restrict__ vtb, bf16_t* __restrict__ yb) {
  constexpr int T = 2048, C = 2048, H = 16, HD = 128, WINDOW = 512, SINK = 4;
  constexpr float SCALE = 0.08838834764831845f;  // 1/sqrt(128)

  const int blk = blockIdx.x;
  const int qblk = blk % (T / 128);
  const int h    = (blk / (T / 128)) % H;
  const int b    = blk / ((T / 128) * H);

  const int wave  = threadIdx.x >> 5;
  const int lane  = threadIdx.x & 31;
  const int lrow  = lane & 15;
  const int lhalf = lane >> 4;
  const int qlo   = qblk * 128 + wave * 16;

  __shared__ __align__(16) bf16_t pstage[8][16 * 32];  // per-wave P staging
  bf16_t* myp = &pstage[wave][0];

  // Preload Q A-fragments for all 4 K-steps of hd=128 (reused every chunk)
  const bf16_t* qrow = qb + ((size_t)(b * T + qlo + lrow)) * C + h * HD + lhalf * 16;
  v16bf aq[4];
#pragma unroll
  for (int kk = 0; kk < 4; ++kk) aq[kk] = frag_contig(qrow + kk * 32);

  v8f o[8];
#pragma unroll
  for (int t = 0; t < 8; ++t)
#pragma unroll
    for (int r = 0; r < 8; ++r) o[t][r] = 0.0f;
  float mrow[8], lsum[8];
#pragma unroll
  for (int r = 0; r < 8; ++r) { mrow[r] = -1e30f; lsum[r] = 0.0f; }

  const int wlo   = qlo - (WINDOW - 1);
  const int first = (wlo > 0 ? wlo : 0) >> 5;
  const int last  = (qlo + 15) >> 5;
  const int nwin  = last - first + 1;
  const int nch   = nwin + (first > 0 ? 1 : 0);  // extra sink chunk

  for (int ci = 0; ci < nch; ++ci) {
    const int c   = (ci < nwin) ? (first + ci) : 0;
    const int kb0 = c * 32;

    // --- scores S(16x32) = Q @ K^T ---
    v8f s[2];
#pragma unroll
    for (int nt = 0; nt < 2; ++nt)
#pragma unroll
      for (int r = 0; r < 8; ++r) s[nt][r] = 0.0f;

    const bf16_t* krow0 =
        kb + ((size_t)(b * T + kb0 + lrow)) * C + h * HD + lhalf * 16;
#pragma unroll
    for (int kk = 0; kk < 4; ++kk) {
      v16bf bk0 = frag_contig(krow0 + kk * 32);
      v16bf bk1 = frag_contig(krow0 + (size_t)16 * C + kk * 32);
      s[0] = wmma_bf16(aq[kk], bk0, s[0]);
      s[1] = wmma_bf16(aq[kk], bk1, s[1]);
    }

    // --- scale + mask (exact, per element) ---
#pragma unroll
    for (int nt = 0; nt < 2; ++nt) {
#pragma unroll
      for (int r = 0; r < 8; ++r) {
        const int i = qlo + r + lhalf * 8;
        const int j = kb0 + nt * 16 + lrow;
        const bool ok = (j <= i) && ((j >= i - (WINDOW - 1)) || (j < SINK));
        s[nt][r] = ok ? s[nt][r] * SCALE : -1e30f;
      }
    }

    // --- online softmax (row stats live per (r, lane-half)) ---
    float alpha[8];
#pragma unroll
    for (int r = 0; r < 8; ++r) {
      float mx = fmaxf(s[0][r], s[1][r]);
      mx = fmaxf(mx, __shfl_xor(mx, 1, 32));
      mx = fmaxf(mx, __shfl_xor(mx, 2, 32));
      mx = fmaxf(mx, __shfl_xor(mx, 4, 32));
      mx = fmaxf(mx, __shfl_xor(mx, 8, 32));
      const float mnew = fmaxf(mrow[r], mx);
      alpha[r] = __expf(mrow[r] - mnew);
      mrow[r]  = mnew;
      s[0][r] = __expf(s[0][r] - mnew);
      s[1][r] = __expf(s[1][r] - mnew);
      float sm = s[0][r] + s[1][r];
      sm += __shfl_xor(sm, 1, 32);
      sm += __shfl_xor(sm, 2, 32);
      sm += __shfl_xor(sm, 4, 32);
      sm += __shfl_xor(sm, 8, 32);
      lsum[r] = lsum[r] * alpha[r] + sm;
    }
#pragma unroll
    for (int t = 0; t < 8; ++t)
#pragma unroll
      for (int r = 0; r < 8; ++r) o[t][r] *= alpha[r];

    // --- P(16x32) f32 C-layout -> bf16 A-layout via wave-private LDS ---
#pragma unroll
    for (int nt = 0; nt < 2; ++nt)
#pragma unroll
      for (int r = 0; r < 8; ++r)
        myp[(r + lhalf * 8) * 32 + nt * 16 + lrow] = (bf16_t)s[nt][r];
    asm volatile("s_wait_dscnt 0" ::: "memory");
    v16bf pfrag = frag_contig(myp + lrow * 32 + lhalf * 16);

    // --- O += P @ V  (V^T layout makes B-columns contiguous along keys) ---
    const bf16_t* vcol =
        vtb + ((size_t)(b * C + h * HD + lrow)) * T + kb0 + lhalf * 16;
#pragma unroll
    for (int t = 0; t < 8; ++t) {
      v16bf bv = frag_contig(vcol + (size_t)(t * 16) * T);
      o[t] = wmma_bf16(pfrag, bv, o[t]);
    }
  }

  // --- epilogue: normalize and store bf16 (B,T,C) ---
#pragma unroll
  for (int t = 0; t < 8; ++t) {
#pragma unroll
    for (int r = 0; r < 8; ++r) {
      const int i   = qlo + r + lhalf * 8;
      const int col = h * HD + t * 16 + lrow;
      yb[((size_t)(b * T + i)) * C + col] = (bf16_t)(o[t][r] / lsum[r]);
    }
  }
}

// ---------------------------------------------------------------------------
// Host launch
// ---------------------------------------------------------------------------
extern "C" void kernel_launch(void* const* d_in, const int* in_sizes, int n_in,
                              void* d_out, int out_size, void* d_ws,
                              size_t ws_size, hipStream_t stream) {
  constexpr int B = 2, T = 2048, C = 2048, L = 512, H = 16;
  constexpr int M = B * T;
  (void)in_sizes; (void)n_in; (void)out_size; (void)ws_size;

  const float* x   = (const float*)d_in[0];
  const float* Wq  = (const float*)d_in[1];
  const float* Wkv = (const float*)d_in[2];
  const float* Wku = (const float*)d_in[3];
  const float* Wvu = (const float*)d_in[4];
  const float* Wo  = (const float*)d_in[5];

  size_t off = 0;
  auto alloc = [&](size_t elems) {
    bf16_t* p = (bf16_t*)((char*)d_ws + off);
    off += (elems * sizeof(bf16_t) + 255) & ~(size_t)255;
    return p;
  };
  bf16_t* xb   = alloc((size_t)M * C);
  bf16_t* Wqb  = alloc((size_t)C * C);
  bf16_t* Wkvb = alloc((size_t)L * C);
  bf16_t* Wkub = alloc((size_t)C * L);
  bf16_t* Wvub = alloc((size_t)C * L);
  bf16_t* Wob  = alloc((size_t)C * C);
  bf16_t* qb   = alloc((size_t)M * C);
  bf16_t* latb = alloc((size_t)M * L);
  bf16_t* kbuf = alloc((size_t)M * C);
  bf16_t* vtb  = alloc((size_t)M * C);  // (B, C, T) transposed V
  bf16_t* yb   = alloc((size_t)M * C);

  auto cvt = [&](const float* src, bf16_t* dst, size_t n) {
    const int blocks = (int)((n / 4 + 255) / 256);
    cvt_f32_bf16<<<dim3(blocks), dim3(256), 0, stream>>>(src, dst, (int)n);
  };
  cvt(x,   xb,   (size_t)M * C);
  cvt(Wq,  Wqb,  (size_t)C * C);
  cvt(Wkv, Wkvb, (size_t)L * C);
  cvt(Wku, Wkub, (size_t)C * L);
  cvt(Wvu, Wvub, (size_t)C * L);
  cvt(Wo,  Wob,  (size_t)C * C);

  // q = x @ Wq^T                      (M,C) x (C,C)^T
  gemm_tn<0><<<dim3(M / 64, C / 256), 256, 0, stream>>>(xb, Wqb, qb, M, C, C, 0);
  // latent = x @ Wkv_down^T           (M,C) x (L,C)^T
  gemm_tn<0><<<dim3(M / 64, L / 256), 256, 0, stream>>>(xb, Wkvb, latb, M, L, C, 0);
  // k = latent @ Wk_up^T              (M,L) x (C,L)^T
  gemm_tn<0><<<dim3(M / 64, C / 256), 256, 0, stream>>>(latb, Wkub, kbuf, M, C, L, 0);
  // v^T = (latent @ Wv_up^T)^T  -> stored (B, C, T)
  gemm_tn<2><<<dim3(M / 64, C / 256), 256, 0, stream>>>(latb, Wvub, vtb, M, C, L, T);
  // attention
  attn_kernel<<<dim3(B * H * (T / 128)), 256, 0, stream>>>(qb, kbuf, vtb, yb);
  // out = y @ Wo^T (f32 straight to d_out)
  gemm_tn<1><<<dim3(M / 64, C / 256), 256, 0, stream>>>(yb, Wob, d_out, M, C, C, 0);
}
